// ExpertsChooseMaskedExpand_25348896981197
// MI455X (gfx1250) — matrix operand order repaired and checked by reading, not compiled, not run on hardware
//
#include <hip/hip_runtime.h>

// Problem constants (from reference)
#define B_ 4
#define T_ 4096
#define D_ 2048
#define E_ 8
#define C_ 512
#define O_ 2048
#define I_ 256  // D_/E_

typedef __attribute__((ext_vector_type(16))) __bf16         bf16x16;
typedef __attribute__((ext_vector_type(2)))  __bf16         bf16x2;
typedef __attribute__((ext_vector_type(8)))  float          v8f;
typedef __attribute__((ext_vector_type(8)))  unsigned short ushort8;
typedef __attribute__((ext_vector_type(4)))  unsigned short ushort4v;
typedef __attribute__((ext_vector_type(2)))  unsigned int   uint2v;

// ---- f32 -> bf16 conversions ----
__device__ __forceinline__ unsigned int pack2_bf16(float x, float y) {
#if __has_builtin(__builtin_amdgcn_cvt_pk_bf16_f32)
    union { bf16x2 v; unsigned int u; } c;
    c.v = __builtin_amdgcn_cvt_pk_bf16_f32(x, y);   // 1 VALU, RNE in HW
    return c.u;
#else
    union { float f; unsigned int u; } a, b;
    a.f = x; b.f = y;
    const unsigned int ua = a.u + 0x8000u;          // round-to-nearest
    const unsigned int ub = b.u + 0x8000u;
    return __builtin_amdgcn_perm(ub, ua, 0x07060302u);  // {hi16(ub), hi16(ua)}
#endif
}
__device__ __forceinline__ unsigned short to_bf16_1(float f) {
    union { float f; unsigned int u; } c; c.f = f;
    return (unsigned short)((c.u + 0x8000u) >> 16); // round-to-nearest
}

__device__ __forceinline__ void cvt4_store(unsigned short* dst, float4 v) {
    uint2v p;
    p[0] = pack2_bf16(v.x, v.y);
    p[1] = pack2_bf16(v.z, v.w);
    *(uint2v*)dst = p;   // single 8-byte ds_store
}

union Frag { bf16x16 v; ushort8 h[2]; };

// Generic batched GEMM: C[m,n] = sum_k A[m,k]*B[k,n] (+ bias[n])
// TA/TB/TC: float (f32) or unsigned short (bf16 raw bits)
// A addr = Ab + m*a_sm + k*a_sk   (A_KC: a_sk==1, else a_sm==1)
// B addr = Bb + k*b_sk + n*b_sn   (B_NC: b_sn==1, else b_sk==1)
template<class TA, class TB, class TC, bool A_KC, bool B_NC>
__global__ __launch_bounds__(256) void moe_gemm_bf16_wmma(
    const TA* __restrict__ A, const TB* __restrict__ B,
    TC* __restrict__ C, const float* __restrict__ bias,
    int K,
    long long a_sm, long long a_sk,
    long long b_sk, long long b_sn,
    long long c_sm,
    int inner,
    long long a_bo, long long a_bi,
    long long b_bo, long long b_bi,
    long long c_bo, long long c_bi)
{
    constexpr int BM = 128, BN = 128, TK = 32;
    constexpr int P = TK + 8;           // 40 u16 = 80B pitch (16B multiple)

    __shared__ __align__(16) unsigned short As[BM][P];   // [m][k]
    __shared__ __align__(16) unsigned short Bs[BN][P];   // transposed: [n][k]

    const int tid  = threadIdx.x;
    const int lane = tid & 31;
    const int wave = tid >> 5;
    const int wm   = wave & 3;    // 4 waves along M (32 rows each)
    const int wn   = wave >> 2;   // 2 waves along N (64 cols each)
    const int hf   = lane >> 4;
    const int l16  = lane & 15;

    const int bz = blockIdx.z;
    const int bo = bz / inner, bi = bz % inner;
    const TA* Ab = A + (long long)bo * a_bo + (long long)bi * a_bi;
    const TB* Bb = B + (long long)bo * b_bo + (long long)bi * b_bi;
    TC*       Cb = C + (long long)bo * c_bo + (long long)bi * c_bi;

    const long long m0 = (long long)blockIdx.y * BM;
    const long long n0 = (long long)blockIdx.x * BN;

    // register staging (global -> regs -> LDS pipeline)
    float4   raf[4], rbf[4];
    ushort8  rau[2];
    ushort4v rbu[4];

    auto load_a = [&](int k0) {
        if constexpr (sizeof(TA) == 4) {
            if constexpr (A_KC) {
                const int kk = (tid & 7) * 4, mb = tid >> 3;
                #pragma unroll
                for (int j = 0; j < 4; ++j)
                    raf[j] = *(const float4*)(Ab + (m0 + mb + 32 * j) * a_sm + (k0 + kk));
            } else {        // m-contiguous (a_sm == 1)
                const int m4 = (tid & 31) * 4, kb = tid >> 5;
                #pragma unroll
                for (int j = 0; j < 4; ++j)
                    raf[j] = *(const float4*)(Ab + (long long)(k0 + kb + 8 * j) * a_sk + (m0 + m4));
            }
        } else {            // bf16, k-contiguous only
            const int kk = (tid & 3) * 8, mb = tid >> 2;
            #pragma unroll
            for (int j = 0; j < 2; ++j)
                rau[j] = *(const ushort8*)(Ab + (m0 + mb + 64 * j) * a_sm + (k0 + kk));
        }
    };
    auto store_a = [&]() {
        if constexpr (sizeof(TA) == 4) {
            if constexpr (A_KC) {
                const int kk = (tid & 7) * 4, mb = tid >> 3;
                #pragma unroll
                for (int j = 0; j < 4; ++j)
                    cvt4_store(&As[mb + 32 * j][kk], raf[j]);
            } else {
                const int m4 = (tid & 31) * 4, kb = tid >> 5;
                #pragma unroll
                for (int j = 0; j < 4; ++j) {
                    const int k = kb + 8 * j;
                    As[m4 + 0][k] = to_bf16_1(raf[j].x);
                    As[m4 + 1][k] = to_bf16_1(raf[j].y);
                    As[m4 + 2][k] = to_bf16_1(raf[j].z);
                    As[m4 + 3][k] = to_bf16_1(raf[j].w);
                }
            }
        } else {
            const int kk = (tid & 3) * 8, mb = tid >> 2;
            #pragma unroll
            for (int j = 0; j < 2; ++j)
                *(ushort8*)&As[mb + 64 * j][kk] = rau[j];   // direct 16B copy
        }
    };
    auto load_b = [&](int k0) {
        if constexpr (sizeof(TB) == 4) {
            if constexpr (B_NC) {
                const int nn = (tid & 31) * 4, kb = tid >> 5;
                #pragma unroll
                for (int j = 0; j < 4; ++j)
                    rbf[j] = *(const float4*)(Bb + (long long)(k0 + kb + 8 * j) * b_sk + (n0 + nn));
            } else {        // k-contiguous
                const int kk = (tid & 7) * 4, nb = tid >> 3;
                #pragma unroll
                for (int j = 0; j < 4; ++j)
                    rbf[j] = *(const float4*)(Bb + (n0 + nb + 32 * j) * b_sn + (k0 + kk));
            }
        } else {            // bf16, n-contiguous only
            const int nn = (tid & 31) * 4, kb = tid >> 5;
            #pragma unroll
            for (int j = 0; j < 4; ++j)
                rbu[j] = *(const ushort4v*)(Bb + (long long)(k0 + kb + 8 * j) * b_sk + (n0 + nn));
        }
    };
    auto store_b = [&]() {
        if constexpr (sizeof(TB) == 4) {
            if constexpr (B_NC) {   // global [k][n] -> LDS [n][k] transpose scatter
                const int nn = (tid & 31) * 4, kb = tid >> 5;
                #pragma unroll
                for (int j = 0; j < 4; ++j) {
                    const int k = kb + 8 * j;
                    Bs[nn + 0][k] = to_bf16_1(rbf[j].x);
                    Bs[nn + 1][k] = to_bf16_1(rbf[j].y);
                    Bs[nn + 2][k] = to_bf16_1(rbf[j].z);
                    Bs[nn + 3][k] = to_bf16_1(rbf[j].w);
                }
            } else {                // [n][k] -> [n][k], packed 8B stores
                const int kk = (tid & 7) * 4, nb = tid >> 3;
                #pragma unroll
                for (int j = 0; j < 4; ++j)
                    cvt4_store(&Bs[nb + 32 * j][kk], rbf[j]);
            }
        } else {            // bf16 transpose scatter, no conversion
            const int nn = (tid & 31) * 4, kb = tid >> 5;
            #pragma unroll
            for (int j = 0; j < 4; ++j) {
                const int k = kb + 8 * j;
                Bs[nn + 0][k] = rbu[j][0];
                Bs[nn + 1][k] = rbu[j][1];
                Bs[nn + 2][k] = rbu[j][2];
                Bs[nn + 3][k] = rbu[j][3];
            }
        }
    };

    v8f acc[2][4] = {};

    load_a(0);
    load_b(0);

    for (int k0 = 0; k0 < K; k0 += TK) {
        store_a();
        store_b();
        __syncthreads();

        if (k0 + TK < K) {     // prefetch next K-step; overlaps with WMMA below
            load_a(k0 + TK);
            load_b(k0 + TK);
        }

        // A fragment (16-bit 16x32): elems 0..7 <- K=8*hf.., elems 8..15 <- K=16+8*hf..
        Frag a[2];
        #pragma unroll
        for (int i = 0; i < 2; ++i) {
            const int r = wm * 32 + i * 16 + l16;
            a[i].h[0] = *(const ushort8*)&As[r][hf * 8];
            a[i].h[1] = *(const ushort8*)&As[r][16 + hf * 8];
        }
        // B fragment (32x16): per lane col n, elems 0..7 <- K=8*hf.., elems 8..15 <- K=16+8*hf..
        Frag b[4];
        #pragma unroll
        for (int j = 0; j < 4; ++j) {
            const int nc = wn * 64 + j * 16 + l16;
            b[j].h[0] = *(const ushort8*)&Bs[nc][hf * 8];
            b[j].h[1] = *(const ushort8*)&Bs[nc][16 + hf * 8];
        }

        #pragma unroll
        for (int i = 0; i < 2; ++i)
            #pragma unroll
            for (int j = 0; j < 4; ++j)
                acc[i][j] = __builtin_amdgcn_wmma_f32_16x16x32_bf16(
                    false, a[i].v, false, b[j].v, (short)0, acc[i][j], false, false);

        __syncthreads();
    }

    // store C, fuse bias; TC selects f32 or bf16 output
    #pragma unroll
    for (int j = 0; j < 4; ++j) {
        const long long col = n0 + wn * 64 + j * 16 + l16;
        const float badd = bias ? bias[col] : 0.0f;
        #pragma unroll
        for (int i = 0; i < 2; ++i) {
            #pragma unroll
            for (int r = 0; r < 8; ++r) {
                const long long row = m0 + wm * 32 + i * 16 + hf * 8 + r;
                const float v = acc[i][j][r] + badd;
                if constexpr (sizeof(TC) == 4)
                    Cb[row * c_sm + col] = v;
                else
                    Cb[row * c_sm + col] = to_bf16_1(v);
            }
        }
    }
}

extern "C" void kernel_launch(void* const* d_in, const int* in_sizes, int n_in,
                              void* d_out, int out_size, void* d_ws, size_t ws_size,
                              hipStream_t stream) {
    const float* x        = (const float*)d_in[0];  // [B,T,D]
    const float* combine  = (const float*)d_in[1];  // [B,T,E,C]
    const float* dispatch = (const float*)d_in[2];  // [B,T,E,C]
    const float* weight   = (const float*)d_in[3];  // [E,O,I]
    const float* bias     = (const float*)d_in[4];  // [O]
    float* out = (float*)d_out;                     // [B,T,O]

    // bf16 intermediates in workspace
    unsigned short* xd = (unsigned short*)d_ws;                         // [B,E,C,I] bf16
    unsigned short* y  = xd + (size_t)B_ * E_ * C_ * I_;                // [B,E,C,O] bf16

    const long long TEC = (long long)T_ * E_ * C_;
    const long long ECI = (long long)E_ * C_ * I_;
    const long long ECO = (long long)E_ * C_ * O_;

    dim3 blk(256);

    // GEMM 1 (dispatch): per (b,e): xd[C x I] = dispatch[b,:,e,:]^T (C x T) @ x[b,:,e-slice] (T x I)
    // A: f32 m-contig; B: f32 n-contig; C: bf16
    moe_gemm_bf16_wmma<float, float, unsigned short, false, true>
        <<<dim3(I_ / 128, C_ / 128, B_ * E_), blk, 0, stream>>>(
        dispatch, x, xd, nullptr, /*K=*/T_,
        /*a_sm=*/1, /*a_sk=*/(long long)E_ * C_,
        /*b_sk=*/D_, /*b_sn=*/1,
        /*c_sm=*/I_,
        /*inner=*/E_,
        /*a_bo=*/TEC, /*a_bi=*/C_,
        /*b_bo=*/(long long)T_ * D_, /*b_bi=*/I_,
        /*c_bo=*/ECI, /*c_bi=*/(long long)C_ * I_);

    // GEMM 2 (expert FFN): per (b,e): y[C x O] = xd (C x I) @ w[e]^T (I x O) + bias
    // A: bf16 k-contig (direct copy); B: f32 k-contig (weight [O][I]); C: bf16
    moe_gemm_bf16_wmma<unsigned short, float, unsigned short, true, false>
        <<<dim3(O_ / 128, C_ / 128, B_ * E_), blk, 0, stream>>>(
        xd, weight, y, bias, /*K=*/I_,
        /*a_sm=*/I_, /*a_sk=*/1,
        /*b_sk=*/1, /*b_sn=*/I_,
        /*c_sm=*/O_,
        /*inner=*/E_,
        /*a_bo=*/ECI, /*a_bi=*/(long long)C_ * I_,
        /*b_bo=*/0, /*b_bi=*/(long long)O_ * I_,
        /*c_bo=*/ECO, /*c_bi=*/(long long)C_ * O_);

    // GEMM 3 (combine): per b: out[T x O] = combine[b] (T x EC) @ y[b] (EC x O)
    // A: f32 k-contig (packed cvt); B: bf16 n-contig (pure copy); C: f32
    moe_gemm_bf16_wmma<float, unsigned short, float, true, true>
        <<<dim3(O_ / 128, T_ / 128, B_), blk, 0, stream>>>(
        combine, y, out, nullptr, /*K=*/E_ * C_,
        /*a_sm=*/(long long)E_ * C_, /*a_sk=*/1,
        /*b_sk=*/O_, /*b_sn=*/1,
        /*c_sm=*/O_,
        /*inner=*/1,
        /*a_bo=*/TEC, /*a_bi=*/0,
        /*b_bo=*/ECO, /*b_bi=*/0,
        /*c_bo=*/(long long)T_ * O_, /*c_bi=*/0);
}
